// ReadAttention_39573828666316
// MI455X (gfx1250) — matrix operand order repaired and checked by reading, not compiled, not run on hardware
//
#include <hip/hip_runtime.h>

// ---------------------------------------------------------------------------
// ReadAttention for MI455X (gfx1250, wave32)
//   keys  = rs @ Wk.T + bk                  [256 x 256]   (WMMA bf16)
//   sc2   = relu(rs @ Wb.T + bb) / ||keys|| * log2(e)     [1024]
//   out   = softmax_m( sc2 * (keys @ mem.T) )  in log2 domain, [1024 x 65536]
// Output write (256 MB @ 23.3 TB/s ~ 11 us) is the roofline. GEMMs run as
// v_wmma_f32_16x16x32_bf16; softmax is a recompute-based 2-pass scheme so the
// 67M-element dots tensor never round-trips HBM. All exponentials are exp2
// with log2(e) pre-folded into the per-row scale (1 FMA + 1 v_exp per elem).
// ---------------------------------------------------------------------------

typedef __attribute__((ext_vector_type(8)))  float  v8f;
typedef __attribute__((ext_vector_type(16))) __bf16 v16bf;

#define B_    256
#define R_    2048
#define M_    65536
#define L_    64
#define H_    4
#define BH_   1024
#define MTILE 256      // memory rows per workgroup tile
#define LROW  72       // LDS row stride in bf16 elems (144 B -> conflict-free b128)
#define LOG2E 1.4426950408889634f

union Frag16 { uint4 q[2]; v16bf v; };

static __device__ __forceinline__ unsigned pkbf(float a, float b) {
  union { __bf16 h[2]; unsigned u; } x;
  x.h[0] = (__bf16)a; x.h[1] = (__bf16)b;
  return x.u;
}

static __device__ __forceinline__ float exp2g(float x) {
  return __builtin_amdgcn_exp2f(x);           // raw v_exp_f32
}

static __device__ __forceinline__ v8f wmma_bf16(v16bf a, v16bf b, v8f c) {
  // v_wmma_f32_16x16x32_bf16: D = A(16x32) * B(32x16) + C(16x16 f32)
  return __builtin_amdgcn_wmma_f32_16x16x32_bf16(false, a, false, b, (short)0, c,
                                                 false, false);
}

// --------------------------------------------------------------------------
// K1: keys projection, one wave per 16x16 output tile, K=2048 in bf16 WMMA.
// A[b][r] = rs (row major), B[r][o] = Wk[o][r] (Wk row-major = column of B).
// --------------------------------------------------------------------------
__global__ __launch_bounds__(32)
void k_proj(const float* __restrict__ rs, const float* __restrict__ Wk,
            const float* __restrict__ bk, __bf16* __restrict__ keys)
{
  const int lane = threadIdx.x & 31;
  const int half = lane >> 4;
  const int l15  = lane & 15;
  const int b0 = (blockIdx.x >> 4) * 16;
  const int o0 = (blockIdx.x & 15) * 16;

  // A: row b0+l15, K 0..15 (lanes 0-15) / 16..31 (lanes 16-31)
  const float* arow = rs + (size_t)(b0 + l15) * R_ + half * 16;
  // B: column o0+l15 = Wk row, K 0..15 (lanes 0-15) / 16..31 (lanes 16-31)
  const float* brow = Wk + (size_t)(o0 + l15) * R_ + half * 16;

  v8f acc = {0,0,0,0,0,0,0,0};
  for (int kk = 0; kk < R_; kk += 32) {
    const float4* ap0 = (const float4*)(arow + kk);
    const float4* ap1 = (const float4*)(arow + kk + 8);
    float4 x0 = ap0[0], x1 = ap0[1];
    float4 y0 = ap1[0], y1 = ap1[1];
    Frag16 fa;
    fa.q[0] = make_uint4(pkbf(x0.x,x0.y), pkbf(x0.z,x0.w),
                         pkbf(x1.x,x1.y), pkbf(x1.z,x1.w));
    fa.q[1] = make_uint4(pkbf(y0.x,y0.y), pkbf(y0.z,y0.w),
                         pkbf(y1.x,y1.y), pkbf(y1.z,y1.w));
    const float4* bp = (const float4*)(brow + kk);
    float4 z0 = bp[0], z1 = bp[1], z2 = bp[2], z3 = bp[3];
    Frag16 fb;
    fb.q[0] = make_uint4(pkbf(z0.x,z0.y), pkbf(z0.z,z0.w),
                         pkbf(z1.x,z1.y), pkbf(z1.z,z1.w));
    fb.q[1] = make_uint4(pkbf(z2.x,z2.y), pkbf(z2.z,z2.w),
                         pkbf(z3.x,z3.y), pkbf(z3.z,z3.w));
    acc = wmma_bf16(fa.v, fb.v, acc);
  }

  const int col = o0 + l15;
  const float bias = bk[col];
  #pragma unroll
  for (int v = 0; v < 8; ++v) {
    int b = b0 + half * 8 + v;                 // C/D layout: VGPR v -> row v(+8)
    keys[(size_t)b * (H_ * L_) + col] = (__bf16)(acc[v] + bias);
  }
}

// --------------------------------------------------------------------------
// K1b: sc2[bh] = relu(rs[b].Wb[h] + bb[h]) * rsqrt(sum(keys[bh]^2)) * log2(e)
// --------------------------------------------------------------------------
__global__ __launch_bounds__(256)
void k_scale(const float* __restrict__ rs, const float* __restrict__ Wb,
             const float* __restrict__ bb, const __bf16* __restrict__ keys,
             float* __restrict__ scale)
{
  const int bh = blockIdx.x * 256 + threadIdx.x;
  const int b = bh >> 2, h = bh & 3;
  const float4* x = (const float4*)(rs + (size_t)b * R_);
  const float4* w = (const float4*)(Wb + (size_t)h * R_);
  float acc = 0.f;
  for (int r = 0; r < R_ / 4; ++r) {
    float4 xv = x[r], wv = w[r];
    acc += xv.x * wv.x + xv.y * wv.y + xv.z * wv.z + xv.w * wv.w;
  }
  float beta = fmaxf(acc + bb[h], 0.f);
  const __bf16* kr = keys + (size_t)bh * L_;
  float ss = 0.f;
  #pragma unroll
  for (int l = 0; l < L_; ++l) { float kv = (float)kr[l]; ss += kv * kv; }
  scale[bh] = beta * rsqrtf(ss) * LOG2E;
}

// --------------------------------------------------------------------------
// K2 / K4: dots tile GEMM + online softmax (log2 domain).
// Grid: (M_/MTILE, BH_/128), block 256 = 8 waves; wave w owns 16 bh-rows.
// WRITE=false: emit per-(mtile,row) partial (max2, sum of exp2(t2-max2)).
// WRITE=true : recompute dots, write exp2(fma(sc2,acc,-gmax2)) * ginv.
// --------------------------------------------------------------------------
template <bool WRITE>
__global__ __launch_bounds__(256)
void k_attn(const float* __restrict__ memf, const __bf16* __restrict__ keys,
            const float* __restrict__ scale, const float* __restrict__ gmax,
            const float* __restrict__ ginv, float* __restrict__ pmax,
            float* __restrict__ psum, float* __restrict__ out)
{
  __shared__ __align__(16) unsigned short smem_u[MTILE * LROW];
  __bf16* smem = (__bf16*)smem_u;

  // Stage 256x64 memory tile (f32 -> bf16) into LDS, padded rows.
  {
    const int t = threadIdx.x;
    const size_t gm = (size_t)blockIdx.x * MTILE + t;
    const float4* src = (const float4*)(memf + gm * L_);
    unsigned* dst = (unsigned*)(smem_u + t * LROW);
    #pragma unroll
    for (int i = 0; i < 16; ++i) {
      float4 v = src[i];
      dst[i * 2 + 0] = pkbf(v.x, v.y);
      dst[i * 2 + 1] = pkbf(v.z, v.w);
    }
  }
  __syncthreads();

  const int lane = threadIdx.x & 31;
  const int wave = threadIdx.x >> 5;
  const int half = lane >> 4;
  const int l15  = lane & 15;
  const int bh0  = blockIdx.y * 128 + wave * 16;

  // A fragments (keys rows bh0..bh0+15, K=0..63) held in VGPRs for all tiles.
  const __bf16* krow = keys + (size_t)(bh0 + l15) * L_;
  const int ab = half ? 8 : 0;
  Frag16 fa0, fa1;
  fa0.q[0] = *(const uint4*)(krow + ab);
  fa0.q[1] = *(const uint4*)(krow + ab + 16);
  fa1.q[0] = *(const uint4*)(krow + 32 + ab);
  fa1.q[1] = *(const uint4*)(krow + 32 + ab + 16);

  float sc[8], ngm[8], gi[8];
  #pragma unroll
  for (int v = 0; v < 8; ++v) {
    int row = bh0 + half * 8 + v;
    sc[v] = scale[row];
    if (WRITE) { ngm[v] = -gmax[row]; gi[v] = ginv[row]; }
  }

  float mx[8], sm[8];
  #pragma unroll
  for (int v = 0; v < 8; ++v) { mx[v] = -INFINITY; sm[v] = 0.f; }

  const int bb0 = half * 8;
  for (int ms = 0; ms < 16; ++ms) {
    const __bf16* brow = smem + (ms * 16 + l15) * LROW;
    Frag16 fb0, fb1;
    fb0.q[0] = *(const uint4*)(brow + bb0);
    fb0.q[1] = *(const uint4*)(brow + bb0 + 16);
    fb1.q[0] = *(const uint4*)(brow + 32 + bb0);
    fb1.q[1] = *(const uint4*)(brow + 32 + bb0 + 16);

    v8f acc = {0,0,0,0,0,0,0,0};
    acc = wmma_bf16(fa0.v, fb0.v, acc);   // K = 0..31
    acc = wmma_bf16(fa1.v, fb1.v, acc);   // K = 32..63

    if (WRITE) {
      const size_t col = (size_t)blockIdx.x * MTILE + ms * 16 + l15;
      #pragma unroll
      for (int v = 0; v < 8; ++v) {
        float r = exp2g(fmaf(sc[v], acc[v], ngm[v])) * gi[v];
        out[(size_t)(bh0 + half * 8 + v) * M_ + col] = r;
      }
    } else {
      #pragma unroll
      for (int v = 0; v < 8; ++v) {
        float t2 = sc[v] * acc[v];
        float nm = fmaxf(mx[v], t2);
        sm[v] = sm[v] * exp2g(mx[v] - nm) + exp2g(t2 - nm);
        mx[v] = nm;
      }
    }
  }

  if (!WRITE) {
    // Reduce (max, sum) across the 16 lanes that share each row.
    #pragma unroll
    for (int v = 0; v < 8; ++v) {
      #pragma unroll
      for (int off = 1; off < 16; off <<= 1) {
        float omx = __shfl_xor(mx[v], off, 32);
        float osm = __shfl_xor(sm[v], off, 32);
        float nm  = fmaxf(mx[v], omx);
        sm[v] = sm[v] * exp2g(mx[v] - nm) + osm * exp2g(omx - nm);
        mx[v] = nm;
      }
    }
    if (l15 == 0) {
      #pragma unroll
      for (int v = 0; v < 8; ++v) {
        int row = bh0 + half * 8 + v;
        pmax[(size_t)blockIdx.x * BH_ + row] = mx[v];
        psum[(size_t)blockIdx.x * BH_ + row] = sm[v];
      }
    }
  }
}

// --------------------------------------------------------------------------
// K3: merge per-mtile partials -> global row max (log2 domain) and 1/sum.
// --------------------------------------------------------------------------
__global__ __launch_bounds__(256)
void k_merge(const float* __restrict__ pmax, const float* __restrict__ psum,
             float* __restrict__ gmax, float* __restrict__ ginv)
{
  const int bh = blockIdx.x * 256 + threadIdx.x;
  float M = -INFINITY, S = 0.f;
  for (int mt = 0; mt < M_ / MTILE; ++mt) {
    float pm = pmax[(size_t)mt * BH_ + bh];
    float ps = psum[(size_t)mt * BH_ + bh];
    float nm = fmaxf(M, pm);
    S = S * exp2g(M - nm) + ps * exp2g(pm - nm);
    M = nm;
  }
  gmax[bh] = M;
  ginv[bh] = 1.f / S;
}

// --------------------------------------------------------------------------
extern "C" void kernel_launch(void* const* d_in, const int* in_sizes, int n_in,
                              void* d_out, int out_size, void* d_ws, size_t ws_size,
                              hipStream_t stream)
{
  const float* rs  = (const float*)d_in[0];   // [256, 2048]
  const float* mem = (const float*)d_in[1];   // [1, 65536, 64]
  const float* Wk  = (const float*)d_in[2];   // [256, 2048]
  const float* bk  = (const float*)d_in[3];   // [256]
  const float* Wb  = (const float*)d_in[4];   // [4, 2048]
  const float* bb  = (const float*)d_in[5];   // [4]
  float* out = (float*)d_out;                 // [1024, 65536]

  char* ws = (char*)d_ws;                     // ~2.25 MB used
  __bf16* keys = (__bf16*)ws;                             // 128 KB [1024][64]
  float*  scale = (float*)(ws + (128 << 10));             // 4 KB
  float*  gmax  = (float*)(ws + (132 << 10));             // 4 KB
  float*  ginv  = (float*)(ws + (136 << 10));             // 4 KB
  float*  pmax  = (float*)(ws + (256 << 10));             // 1 MB
  float*  psum  = (float*)(ws + (256 << 10) + (1 << 20)); // 1 MB

  hipLaunchKernelGGL(k_proj,  dim3(256), dim3(32),  0, stream, rs, Wk, bk, keys);
  hipLaunchKernelGGL(k_scale, dim3(4),   dim3(256), 0, stream, rs, Wb, bb, keys, scale);
  hipLaunchKernelGGL((k_attn<false>), dim3(M_ / MTILE, BH_ / 128), dim3(256), 0, stream,
                     mem, keys, scale, gmax, ginv, pmax, psum, out);
  hipLaunchKernelGGL(k_merge, dim3(4),   dim3(256), 0, stream, pmax, psum, gmax, ginv);
  hipLaunchKernelGGL((k_attn<true>),  dim3(M_ / MTILE, BH_ / 128), dim3(256), 0, stream,
                     mem, keys, scale, gmax, ginv, pmax, psum, out);
}
